// LinearAttentionParallel_17317308138079
// MI455X (gfx1250) — matrix operand order repaired and checked by reading, not compiled, not run on hardware
//
#include <hip/hip_runtime.h>
#include <hip/hip_bf16.h>
#include <cmath>
#include <cstdint>

#define N_EMBD    1024
#define T_SEQ     2048
#define BATCH     8

typedef __attribute__((ext_vector_type(16))) __bf16          v16bf;
typedef __attribute__((ext_vector_type(8)))  float           v8f;
typedef __attribute__((ext_vector_type(8)))  unsigned short  u16x8;

union BFrag {
    struct { u16x8 lo, hi; } h;
    v16bf v;
};

__device__ __forceinline__ unsigned short f2bf(float f) {
    unsigned int u = __builtin_bit_cast(unsigned int, f);
    unsigned int r = u + 0x7FFFu + ((u >> 16) & 1u);   // round-to-nearest-even
    return (unsigned short)(r >> 16);
}

// ---------------------------------------------------------------------------
// CDNA5 async global->LDS copy (GLOBAL_LOAD_ASYNC_TO_LDS_B128, ASYNCcnt).
// Builtin signature (from clang diagnostic): arg0 = AS1 int4*, arg1 = AS3 int4*,
// then imm offset, imm cpol.  Generic->AS casts done via uintptr_t (the low
// 32 bits of a generic LDS pointer are the LDS offset per ISA flat addressing).
// ---------------------------------------------------------------------------
#if __has_builtin(__builtin_amdgcn_global_load_async_to_lds_b128) && \
    __has_builtin(__builtin_amdgcn_s_wait_asynccnt)
#define HAS_ASYNC_LDS 1
#else
#define HAS_ASYNC_LDS 0
#endif

#if HAS_ASYNC_LDS
typedef int v4i_vs __attribute__((vector_size(4 * sizeof(int))));
typedef __attribute__((address_space(1))) v4i_vs GlobalInt4;
typedef __attribute__((address_space(3))) v4i_vs LdsInt4;

__device__ __forceinline__ void cp_async16(const void* g, void* l) {
    __builtin_amdgcn_global_load_async_to_lds_b128(
        (GlobalInt4*)(uintptr_t)g,
        (LdsInt4*)(unsigned int)(uintptr_t)l,
        0, 0);
}
#endif

// ---------------------------------------------------------------------------
// fp32 -> bf16 elementwise conversion
// ---------------------------------------------------------------------------
__global__ void f32_to_bf16_kernel(const float* __restrict__ in,
                                   unsigned short* __restrict__ out, int n) {
    int i = blockIdx.x * blockDim.x + threadIdx.x;
    if (i < n) out[i] = f2bf(in[i]);
}

// ---------------------------------------------------------------------------
// fp32 [R x C] -> bf16 transposed [C x R] (LDS 32x33 tile transpose)
// ---------------------------------------------------------------------------
__global__ __launch_bounds__(256)
void f32_to_bf16_transpose_kernel(const float* __restrict__ in,
                                  unsigned short* __restrict__ out,
                                  int R, int C) {
    __shared__ unsigned short tile[32][33];
    const int c0 = blockIdx.x * 32;
    const int r0 = blockIdx.y * 32;
    const int tx = threadIdx.x & 31;
    const int ty = threadIdx.x >> 5;   // 0..7
    #pragma unroll
    for (int i = 0; i < 4; i++) {
        int r = ty + i * 8;
        tile[r][tx] = f2bf(in[(long long)(r0 + r) * C + (c0 + tx)]);
    }
    __syncthreads();
    #pragma unroll
    for (int i = 0; i < 4; i++) {
        int r = ty + i * 8;            // row within the c-tile of the output
        out[(long long)(c0 + r) * R + (r0 + tx)] = tile[tx][r];
    }
}

// ---------------------------------------------------------------------------
// bf16 WMMA GEMM: C = epilogue(A @ B^T_layout), fp32 accumulate.
//   A  : [M x K] row-major (lda = K)
//   Bt : [N x K] row-major (ldb = K)   (B pre-transposed -> b128 loads)
// 128x128 block tile, 128 threads (4 wave32), each wave 64x64 (4x4 WMMA),
// BK = 32, double-buffered LDS filled by async global->LDS DMA.
// ---------------------------------------------------------------------------
enum { EPI_GELU_BIAS = 0, EPI_BIAS = 1, EPI_BIAS_TRANS = 2, EPI_MASK = 3, EPI_F32 = 4 };

template <int EPI>
__global__ __launch_bounds__(128, 1)
void gemm_wmma_kernel(const unsigned short* __restrict__ A,
                      const unsigned short* __restrict__ Bt,
                      const float* __restrict__ bias,
                      void* __restrict__ Cout,
                      int M, int N, int K,
                      long long strideA, long long strideB, long long strideC,
                      float inv_scale) {
    const int t    = threadIdx.x;
    const int lane = t & 31;
    const int wave = t >> 5;
    const int m0   = blockIdx.y * 128;
    const int n0   = blockIdx.x * 128;

    const unsigned short* Ag = A  + (long long)blockIdx.z * strideA;
    const unsigned short* Bg = Bt + (long long)blockIdx.z * strideB;

    // Fully-masked (strictly upper-triangular) score tiles: zero-fill, skip GEMM.
    if (EPI == EPI_MASK && n0 > m0 + 127) {
        unsigned short* Og = (unsigned short*)Cout + (long long)blockIdx.z * strideC;
        const uint4 z4 = {0u, 0u, 0u, 0u};
        for (int idx = t; idx < 128 * 16; idx += 128) {        // 8 bf16 per store
            int r = idx >> 4, cg = idx & 15;
            *(uint4*)(Og + (long long)(m0 + r) * N + n0 + cg * 8) = z4;
        }
        return;
    }

    __shared__ alignas(16) unsigned short As[2][128][32];      // 16 KB
    __shared__ alignas(16) unsigned short Bs[2][128][32];      // 16 KB

    const int wm  = (wave >> 1) * 64;
    const int wn  = (wave & 1) * 64;
    const int l15 = lane & 15;
    const int akb = (lane < 16) ? 0 : 8;    // A frag: 8 elems @akb, 8 @akb+16
    const int bkb = (lane < 16) ? 0 : 8;    // B frag: 8 elems @bkb, 8 @bkb+16

    const v8f vzero = {0.f, 0.f, 0.f, 0.f, 0.f, 0.f, 0.f, 0.f};
    v8f acc[4][4];
    #pragma unroll
    for (int i = 0; i < 4; i++)
        #pragma unroll
        for (int j = 0; j < 4; j++) acc[i][j] = vzero;

    // Fill one LDS buffer: 8 x b128 per thread (4 for A, 4 for B), no VGPR staging.
    auto fill = [&](int buf, int kb) {
        #pragma unroll
        for (int c = 0; c < 2; c++) {
            const int id = t + c * 128, row = id >> 1, half = id & 1;
            const unsigned short* ga = Ag + (long long)(m0 + row) * K + kb + half * 16;
            const unsigned short* gb = Bg + (long long)(n0 + row) * K + kb + half * 16;
            unsigned short* la = &As[buf][row][half * 16];
            unsigned short* lb = &Bs[buf][row][half * 16];
#if HAS_ASYNC_LDS
            cp_async16(ga,     la);
            cp_async16(ga + 8, la + 8);
            cp_async16(gb,     lb);
            cp_async16(gb + 8, lb + 8);
#else
            uint4 a0 = ((const uint4*)ga)[0], a1 = ((const uint4*)ga)[1];
            uint4 b0 = ((const uint4*)gb)[0], b1 = ((const uint4*)gb)[1];
            ((uint4*)la)[0] = a0; ((uint4*)la)[1] = a1;
            ((uint4*)lb)[0] = b0; ((uint4*)lb)[1] = b1;
#endif
        }
    };

    fill(0, 0);

    const int nk = K / 32;
    for (int ki = 0; ki < nk; ki++) {
        const int cur = ki & 1;
        if (ki + 1 < nk) fill(1 - cur, (ki + 1) * 32);   // overlap next with WMMA
        if (ki + 2 < nk) {                                // L2 prefetch k+2
            __builtin_prefetch(Ag + (long long)(m0 + (t >> 1)) * K + (ki + 2) * 32, 0, 0);
            __builtin_prefetch(Bg + (long long)(n0 + (t >> 1)) * K + (ki + 2) * 32, 0, 0);
        }
#if HAS_ASYNC_LDS
        if (ki + 1 < nk) __builtin_amdgcn_s_wait_asynccnt(8);  // cur landed, next in flight
        else             __builtin_amdgcn_s_wait_asynccnt(0);
#endif
        __syncthreads();   // cur buffer visible to all waves

        BFrag a[4], b[4];
        #pragma unroll
        for (int i = 0; i < 4; i++) {
            const unsigned short* ap = &As[cur][wm + i * 16 + l15][0];
            a[i].h.lo = *(const u16x8*)(ap + akb);
            a[i].h.hi = *(const u16x8*)(ap + akb + 16);
            const unsigned short* bp = &Bs[cur][wn + i * 16 + l15][0];
            b[i].h.lo = *(const u16x8*)(bp + bkb);
            b[i].h.hi = *(const u16x8*)(bp + bkb + 16);
        }
        #pragma unroll
        for (int i = 0; i < 4; i++)
            #pragma unroll
            for (int j = 0; j < 4; j++)
                acc[i][j] = __builtin_amdgcn_wmma_f32_16x16x32_bf16(
                    false, a[i].v, false, b[j].v, (short)0, acc[i][j], false, false);

        __syncthreads();   // all waves done reading cur before it is refilled
    }

    // ---- epilogue (C/D layout: VGPR r -> row r for lanes 0-15, r+8 for 16-31)
    const int rofs = (lane < 16) ? 0 : 8;
    float bj[4];
    if (EPI == EPI_GELU_BIAS || EPI == EPI_BIAS || EPI == EPI_BIAS_TRANS) {
        #pragma unroll
        for (int j = 0; j < 4; j++) bj[j] = bias[n0 + wn + j * 16 + l15];
    }
    #pragma unroll
    for (int i = 0; i < 4; i++) {
        #pragma unroll
        for (int j = 0; j < 4; j++) {
            const int cb = n0 + wn + j * 16 + l15;
            #pragma unroll
            for (int r = 0; r < 8; r++) {
                const int rb = m0 + wm + i * 16 + r + rofs;
                float val = acc[i][j][r];
                if (EPI == EPI_F32) {
                    ((float*)Cout)[(long long)blockIdx.z * strideC +
                                   (long long)rb * N + cb] = val;
                } else if (EPI == EPI_GELU_BIAS) {
                    float hv = val + bj[j];
                    hv = 0.5f * hv * (1.0f + erff(hv * 0.70710678118654752f));
                    ((unsigned short*)Cout)[(long long)rb * N + cb] = f2bf(hv);
                } else if (EPI == EPI_BIAS) {
                    ((unsigned short*)Cout)[(long long)rb * N + cb] = f2bf(val + bj[j]);
                } else if (EPI == EPI_BIAS_TRANS) {
                    // write v transposed per batch: vT[b][c][t], b = rb/T, t = rb%T
                    const long long off =
                        ((long long)(rb >> 11) * N_EMBD + cb) * T_SEQ + (rb & (T_SEQ - 1));
                    ((unsigned short*)Cout)[off] = f2bf(val + bj[j]);
                } else { // EPI_MASK
                    ((unsigned short*)Cout)[(long long)blockIdx.z * strideC +
                                            (long long)rb * N + cb] =
                        (cb <= rb) ? f2bf(val * inv_scale) : (unsigned short)0;
                }
            }
        }
    }
}

// ---------------------------------------------------------------------------
// LayerNorm over last dim (1024), weight+bias, eps=1e-5.  One block per row.
// ---------------------------------------------------------------------------
__global__ __launch_bounds__(256)
void layernorm_kernel(const float* __restrict__ in,
                      const float* __restrict__ w,
                      const float* __restrict__ b,
                      float* __restrict__ out) {
    const int row = blockIdx.x;
    const int t   = threadIdx.x;
    const float* x = in + (long long)row * N_EMBD;
    float* o       = out + (long long)row * N_EMBD;

    float4 xv = ((const float4*)x)[t];
    float s  = xv.x + xv.y + xv.z + xv.w;
    float s2 = xv.x * xv.x + xv.y * xv.y + xv.z * xv.z + xv.w * xv.w;

    __shared__ float2 red[256];
    red[t] = make_float2(s, s2);
    __syncthreads();
    for (int ofs = 128; ofs > 0; ofs >>= 1) {
        if (t < ofs) {
            red[t].x += red[t + ofs].x;
            red[t].y += red[t + ofs].y;
        }
        __syncthreads();
    }
    const float mu   = red[0].x * (1.0f / N_EMBD);
    const float var  = red[0].y * (1.0f / N_EMBD) - mu * mu;
    const float rstd = rsqrtf(var + 1e-5f);

    float4 wv = ((const float4*)w)[t];
    float4 bv = ((const float4*)b)[t];
    float4 ov;
    ov.x = (xv.x - mu) * rstd * wv.x + bv.x;
    ov.y = (xv.y - mu) * rstd * wv.y + bv.y;
    ov.z = (xv.z - mu) * rstd * wv.z + bv.z;
    ov.w = (xv.w - mu) * rstd * wv.w + bv.w;
    ((float4*)o)[t] = ov;
}

// ---------------------------------------------------------------------------
// Host launcher
// ---------------------------------------------------------------------------
extern "C" void kernel_launch(void* const* d_in, const int* in_sizes, int n_in,
                              void* d_out, int out_size, void* d_ws, size_t ws_size,
                              hipStream_t stream) {
    const float* x    = (const float*)d_in[0];
    const float* W1[3] = {(const float*)d_in[1], (const float*)d_in[5], (const float*)d_in[9]};
    const float* b1[3] = {(const float*)d_in[2], (const float*)d_in[6], (const float*)d_in[10]};
    const float* W2[3] = {(const float*)d_in[3], (const float*)d_in[7], (const float*)d_in[11]};
    const float* b2[3] = {(const float*)d_in[4], (const float*)d_in[8], (const float*)d_in[12]};
    const float* ln_w  = (const float*)d_in[13];
    const float* ln_b  = (const float*)d_in[14];

    const long long C   = N_EMBD;
    const long long CC  = C * C;                       // 1,048,576
    const long long BT  = (long long)BATCH * T_SEQ;    // 16,384
    const long long BTC = BT * C;                      // 16,777,216
    const long long TT  = (long long)T_SEQ * T_SEQ;    // 4,194,304

    // workspace carve-up (all offsets are large powers of two -> aligned)
    char* ws = (char*)d_ws;
    unsigned short* x16   = (unsigned short*)ws; ws += BTC * 2;        //  32 MB
    unsigned short* w16t  = (unsigned short*)ws; ws += 6 * CC * 2;     //  12 MB (transposed)
    unsigned short* h16   = (unsigned short*)ws; ws += BTC * 2;        //  32 MB
    unsigned short* q16   = (unsigned short*)ws; ws += BTC * 2;        //  32 MB
    unsigned short* k16   = (unsigned short*)ws; ws += BTC * 2;        //  32 MB
    unsigned short* vT16  = (unsigned short*)ws; ws += BTC * 2;        //  32 MB  [B][C][T]
    unsigned short* s16   = (unsigned short*)ws; ws += BATCH * TT * 2; //  64 MB
    float*          out32 = (float*)ws;                                //  64 MB

    // 1) x -> bf16; weights -> bf16 TRANSPOSED ([N][K] layout for b128 loads)
    f32_to_bf16_kernel<<<dim3((unsigned)((BTC + 255) / 256)), 256, 0, stream>>>(
        x, x16, (int)BTC);
    dim3 gT(N_EMBD / 32, N_EMBD / 32, 1);
    for (int i = 0; i < 3; i++) {
        f32_to_bf16_transpose_kernel<<<gT, 256, 0, stream>>>(
            W1[i], w16t + (2 * i) * CC, N_EMBD, N_EMBD);
        f32_to_bf16_transpose_kernel<<<gT, 256, 0, stream>>>(
            W2[i], w16t + (2 * i + 1) * CC, N_EMBD, N_EMBD);
    }

    // 2) MLPs: h = GELU(x@W1 + b1); y = h@W2 + b2
    //    q,k stored [B,T,C]; v stored transposed [B][C][T] for the scores@v GEMM.
    dim3 gMLP(N_EMBD / 128, (unsigned)(BT / 128), 1);   // 8 x 128
    unsigned short* ydst[3] = {q16, k16, vT16};
    for (int i = 0; i < 3; i++) {
        gemm_wmma_kernel<EPI_GELU_BIAS><<<gMLP, 128, 0, stream>>>(
            x16, w16t + (2 * i) * CC, b1[i], h16,
            (int)BT, N_EMBD, N_EMBD, 0, 0, 0, 0.0f);
        if (i < 2)
            gemm_wmma_kernel<EPI_BIAS><<<gMLP, 128, 0, stream>>>(
                h16, w16t + (2 * i + 1) * CC, b2[i], ydst[i],
                (int)BT, N_EMBD, N_EMBD, 0, 0, 0, 0.0f);
        else
            gemm_wmma_kernel<EPI_BIAS_TRANS><<<gMLP, 128, 0, stream>>>(
                h16, w16t + (2 * i + 1) * CC, b2[i], ydst[i],
                (int)BT, N_EMBD, N_EMBD, 0, 0, 0, 0.0f);
    }

    // 3) scores = tril(q @ k^T) / sqrt(N_EMBD * BLOCK_SIZE)
    //    k[b] is [T][C] row-major == [N][K] layout -> direct b128 loads.
    const float inv_scale = 1.0f / sqrtf((float)N_EMBD * (float)T_SEQ);
    dim3 gS(T_SEQ / 128, T_SEQ / 128, BATCH);           // 16 x 16 x 8
    gemm_wmma_kernel<EPI_MASK><<<gS, 128, 0, stream>>>(
        q16, k16, nullptr, s16,
        T_SEQ, T_SEQ, N_EMBD,
        (long long)T_SEQ * C, (long long)T_SEQ * C, TT, inv_scale);

    // 4) out32 = scores @ v : A = scores [T][T], B = vT[b] [C][T] = [N][K]
    dim3 gO(N_EMBD / 128, T_SEQ / 128, BATCH);          // 8 x 16 x 8
    gemm_wmma_kernel<EPI_F32><<<gO, 128, 0, stream>>>(
        s16, vT16, nullptr, out32,
        T_SEQ, N_EMBD, T_SEQ,
        TT, (long long)T_SEQ * C, (long long)T_SEQ * C, 0.0f);

    // 5) LayerNorm -> d_out (fp32)
    layernorm_kernel<<<dim3((unsigned)BT), 256, 0, stream>>>(
        out32, ln_w, ln_b, (float*)d_out);
}
